// CGRE_68719477510
// MI455X (gfx1250) — matrix operand order repaired and checked by reading, not compiled, not run on hardware
//
#include <hip/hip_runtime.h>
#include <hip/hip_bf16.h>

// ---------------------------------------------------------------------------
// CGRE bag-attention + classifier, restructured as:
//   P[i, 0:53]    = X[i] . W[r]            (classifier projections)
//   P[i, 53:106]  = X[i] . Constraints[r]  (attention scores for all rels)
// via one bf16x3 (hi/lo split) WMMA GEMM reading X exactly once from HBM,
// then a tiny per-bag softmax + weighted reduction.
//
// B = [W; Constraints] is pre-packed once into bf16 hi/lo in the exact WMMA
// B-operand lane layout (931 KB, L2 resident).  A tiles are cooperatively
// staged f32->bf16(hi/lo) through double-buffered LDS (one float2 per thread,
// packed cvt) and shared by all waves.  All 8 waves run WMMA every step
// (wave 7 duplicates tile 0, store suppressed) so the hot loop carries no
// exec-mask toggles and no accumulator copies.
// ---------------------------------------------------------------------------

typedef __attribute__((ext_vector_type(16))) __bf16 v16bf;
typedef __attribute__((ext_vector_type(2)))  __bf16 v2bf;
typedef __attribute__((ext_vector_type(8)))  float  v8f;
typedef __attribute__((ext_vector_type(2)))  float  f32x2;

#define N_SENT   65536
#define N_BAGS   8192
#define D_FEAT   2070
#define N_REL    53
#define N_COLS   106          // rows of M = [W ; Constraints]
#define N_CTILE  7            // ceil(106/16)
#define P_STRIDE 112          // padded column stride of P
#define NKSTEP   65           // ceil(2070/32); last step is the 22-wide tail

// workspace layout (bytes)
#define P_BYTES   ((size_t)N_SENT * P_STRIDE * 4)        // 29,360,128
#define BPK_ELEMS ((size_t)N_CTILE * NKSTEP * 32 * 16)   // 232,960 bf16

// ---------------------------------------------------------------------------
// Pack M = [W;Con] into bf16 hi/lo, B-operand lane layout:
//   linear idx = ((tc*65 + ks)*32 + lane)*16 + i
//   lane<16 : col = tc*16+lane,     K = ks*32 + (i&7) + (i>>3)*16  (K 0..7,16..23)
//   lane>=16: col = tc*16+lane-16,  K = ks*32 + 8 + (i&7) + (i>>3)*16  (K 8..15,24..31)
// ---------------------------------------------------------------------------
__global__ __launch_bounds__(256)
void cgre_pack_B(const float* __restrict__ W, const float* __restrict__ Con,
                 __bf16* __restrict__ Bhi, __bf16* __restrict__ Blo) {
    int idx = blockIdx.x * 256 + threadIdx.x;
    if (idx >= (int)BPK_ELEMS) return;
    int i    = idx & 15;
    int lane = (idx >> 4) & 31;
    int ts   = idx >> 9;          // tc*65 + ks
    int ks   = ts % NKSTEP;
    int tc   = ts / NKSTEP;
    int col  = tc * 16 + (lane & 15);
    int k    = ks * 32 + ((lane >> 4) << 3) + (i & 7) + ((i >> 3) << 4);
    float x = 0.0f;
    if (col < N_COLS && k < D_FEAT)
        x = (col < N_REL) ? W[(size_t)col * D_FEAT + k]
                          : Con[(size_t)(col - N_REL) * D_FEAT + k];
    __bf16 h = (__bf16)x;
    Bhi[idx] = h;
    Blo[idx] = (__bf16)(x - (float)h);
}

// ---------------------------------------------------------------------------
// Kernel: P[65536 x 112] = X @ M^T.  One block per 16-row tile, 8 waves.
// ---------------------------------------------------------------------------
__global__ __launch_bounds__(256)
void cgre_gemm_wmma(const float* __restrict__ X,
                    const __bf16* __restrict__ Bhi,
                    const __bf16* __restrict__ Blo,
                    float* __restrict__ P) {
    __shared__ alignas(32) __bf16 sAhi[2][512];
    __shared__ alignas(32) __bf16 sAlo[2][512];

    const int tid  = threadIdx.x;
    const int lane = tid & 31;
    const int tc   = tid >> 5;                      // 0..7
    const int row0 = blockIdx.x * 16;
    const int tceff = (tc < N_CTILE) ? tc : 0;      // wave 7 duplicates tile 0

    // --- staging geometry: one float2 (K-pair) per thread, constant slot ---
    const int srow = tid >> 4;                      // 0..15
    const int kk   = (tid & 15) << 1;               // 0,2,...,30
    // A-lane layout: lane = half*16 + row ; elem i: K = half*8 + (i&7) + (i>=8)*16
    const int li   = ((((kk >> 3) & 1) << 4) + srow) * 16
                     + (kk & 7) + ((kk >> 4) << 3); // pair (kk,kk+1) -> li,li+1
    const float* xp = X + (size_t)(row0 + srow) * D_FEAT + kk;

    const v16bf* bh = (const v16bf*)Bhi + (size_t)tceff * NKSTEP * 32 + lane;
    const v16bf* bl = (const v16bf*)Blo + (size_t)tceff * NKSTEP * 32 + lane;

    v8f acc0 = {}, acc1 = {};

    auto stage = [&](int k0, int buf, bool guard) {
        f32x2 x = {0.0f, 0.0f};
        if (!guard || (k0 + kk) < D_FEAT)           // D_FEAT even: pair-safe
            x = *(const f32x2*)(xp + k0);
        __bf16 h0 = (__bf16)x.x, h1 = (__bf16)x.y;
        v2bf hp = {h0, h1};
        v2bf lp = {(__bf16)(x.x - (float)h0), (__bf16)(x.y - (float)h1)};
        *(v2bf*)(&sAhi[buf][li]) = hp;
        *(v2bf*)(&sAlo[buf][li]) = lp;
    };

    auto compute = [&](int buf, int ks) {
        v16bf ahi = *(const v16bf*)&sAhi[buf][lane * 16];
        v16bf alo = *(const v16bf*)&sAlo[buf][lane * 16];
        v16bf bhv = bh[(size_t)ks * 32];
        v16bf blv = bl[(size_t)ks * 32];
        acc0 = __builtin_amdgcn_wmma_f32_16x16x32_bf16(false, ahi, false, bhv,
                                                       (short)0, acc0, false, false);
        acc1 = __builtin_amdgcn_wmma_f32_16x16x32_bf16(false, ahi, false, blv,
                                                       (short)0, acc1, false, false);
        acc1 = __builtin_amdgcn_wmma_f32_16x16x32_bf16(false, alo, false, bhv,
                                                       (short)0, acc1, false, false);
    };

    stage(0, 0, false);
    __syncthreads();

    // steady state: 63 clean iterations, no range checks, no exec toggles
    for (int ks = 0; ks < 63; ++ks) {
        stage((ks + 1) * 32, (ks + 1) & 1, false);
        compute(ks & 1, ks);
        __syncthreads();
    }
    // ks = 63: stage guarded tail (k0 = 2048, 22 valid) into buf 0
    stage(64 * 32, 0, true);
    compute(1, 63);
    __syncthreads();
    // ks = 64: tail compute
    compute(0, 64);

    if (tc < N_CTILE) {
        // D layout: VGPR v: lane<16 -> (M=v, N=lane); lane>=16 -> (M=v+8, N=lane-16)
        const int half = lane >> 4, l16 = lane & 15;
        float* pbase = P + (size_t)(row0 + half * 8) * P_STRIDE + (tc * 16 + l16);
#pragma unroll
        for (int v = 0; v < 8; ++v)
            pbase[(size_t)v * P_STRIDE] = acc0[v] + acc1[v];
    }
}

// ---------------------------------------------------------------------------
// Per-bag segment softmax over P[:, 53+rel], weighted sum of P[:, 0:53].
// One wave per bag (bags hold 4..12 sentences).
// ---------------------------------------------------------------------------
__global__ __launch_bounds__(256)
void cgre_bag_reduce(const float* __restrict__ P,
                     const int* __restrict__ Scope,
                     const int* __restrict__ Rel,
                     const float* __restrict__ bias,
                     float* __restrict__ out) {
    const int bag  = (blockIdx.x * blockDim.x + threadIdx.x) >> 5;
    const int lane = threadIdx.x & 31;
    if (bag >= N_BAGS) return;

    const int start = Scope[2 * bag];
    const int end   = Scope[2 * bag + 1];
    const int rel   = Rel[bag];
    const int n     = end - start;
    const int scol  = N_REL + rel;

    float m = -3.402823466e38f;
    for (int j = 0; j < n; ++j)
        m = fmaxf(m, P[(size_t)(start + j) * P_STRIDE + scol]);
    float z = 0.0f;
    for (int j = 0; j < n; ++j)
        z += __expf(P[(size_t)(start + j) * P_STRIDE + scol] - m);
    const float invz = 1.0f / z;

    for (int r = lane; r < N_REL; r += 32) {
        float acc = 0.0f;
        for (int j = 0; j < n; ++j) {
            const size_t base = (size_t)(start + j) * P_STRIDE;
            float w = __expf(P[base + scol] - m) * invz;
            acc += w * P[base + r];
        }
        out[(size_t)bag * N_REL + r] = acc + bias[r];
    }
}

extern "C" void kernel_launch(void* const* d_in, const int* in_sizes, int n_in,
                              void* d_out, int out_size, void* d_ws, size_t ws_size,
                              hipStream_t stream) {
    (void)in_sizes; (void)n_in; (void)out_size; (void)ws_size;
    const float* X     = (const float*)d_in[0];
    const float* Con   = (const float*)d_in[1];
    const float* W     = (const float*)d_in[2];
    const float* bias  = (const float*)d_in[3];
    const int*   Scope = (const int*)d_in[4];
    const int*   Rel   = (const int*)d_in[5];
    float*       out   = (float*)d_out;

    float*  P   = (float*)d_ws;                              // 29.36 MB
    __bf16* Bhi = (__bf16*)((char*)d_ws + P_BYTES);          // 455.0 KB
    __bf16* Blo = Bhi + BPK_ELEMS;                           // 455.0 KB

    cgre_pack_B<<<dim3((unsigned)((BPK_ELEMS + 255) / 256)), dim3(256), 0, stream>>>(
        W, Con, Bhi, Blo);
    cgre_gemm_wmma<<<dim3(N_SENT / 16), dim3(256), 0, stream>>>(X, Bhi, Blo, P);
    cgre_bag_reduce<<<dim3(N_BAGS / 8), dim3(256), 0, stream>>>(P, Scope, Rel, bias, out);
}
